// EdgeClassificationGNN_34359738368130
// MI455X (gfx1250) — compile-verified, matrix operand order checked by reading, not run on hardware
//
#include <hip/hip_runtime.h>
#include <hip/hip_bf16.h>
#include <cmath>

typedef __attribute__((ext_vector_type(16))) _Float16 v16h;
typedef __attribute__((ext_vector_type(8)))  _Float16 v8h;
typedef __attribute__((ext_vector_type(8)))  float    v8f;

#define WAVES_PER_BLOCK 8
#define BLOCK_THREADS   256

// ---------------------------------------------------------------- utilities

__global__ void cvt_f16_kernel(const float* __restrict__ in, _Float16* __restrict__ out, long n) {
  long i = (long)blockIdx.x * blockDim.x + threadIdx.x;
  if (i < n) out[i] = (_Float16)in[i];
}

__global__ void fill_f32_kernel(float* __restrict__ p, float v, long n) {
  long i = (long)blockIdx.x * blockDim.x + threadIdx.x;
  if (i < n) p[i] = v;
}

// Pack a row-major f32 [KDIM x NDIM] weight matrix into WMMA B-fragment order
// (f16). Layout: P[(((kt*(NDIM/16)) + ct)*32 + lane)*16 + j]
//   lane 0-15 : col = ct*16 + lane, K-halves kgrp=0 ; lane 16-31: kgrp=8
//   j 0..7  -> K = kt*32 + kgrp + j
//   j 8..15 -> K = kt*32 + kgrp + 16 + (j-8)
template<int KDIM, int NDIM>
__global__ void pack_b_kernel(const float* __restrict__ B, _Float16* __restrict__ P) {
  long i = (long)blockIdx.x * blockDim.x + threadIdx.x;
  if (i >= (long)KDIM * NDIM) return;
  int  j    = (int)(i & 15);
  long t    = i >> 4;
  int  lane = (int)(t & 31); t >>= 5;
  int  ct   = (int)(t % (NDIM / 16));
  int  kt   = (int)(t / (NDIM / 16));
  int  kgrp = (lane & 16) ? 8 : 0;
  int  k    = kt * 32 + kgrp + ((j < 8) ? j : (8 + j));
  int  col  = ct * 16 + (lane & 15);
  P[i] = (_Float16)B[(long)k * NDIM + col];
}

// -------------------------------------------------- async LDS staging (CDNA5)
// Copies `CHUNKS` b128 chunks of packed-B fragments into LDS using the
// gfx1250 async-to-LDS path (ASYNCcnt), then drains and barriers.
__device__ __forceinline__ void async_copy_b128(uint32_t lds_byte_addr, uint64_t gaddr) {
  asm volatile("global_load_async_to_lds_b128 %0, %1, off"
               :: "v"(lds_byte_addr), "v"(gaddr) : "memory");
}
__device__ __forceinline__ void wait_asynccnt0() {
  asm volatile("s_wait_asynccnt 0x0" ::: "memory");
}

// ---------------------------------------------------------------- WMMA GEMM
// C[M,NDIM] = A16[M,KDIM] @ Bpacked ; one 16-row tile per wave, COLS 16x16
// accumulators per wave. The block's packed-B column group is staged to LDS
// with async copies; A streams from global (LOADcnt) while B comes from LDS
// (DScnt), decoupling the wait chains feeding the WMMAs.
template<int KDIM, int NDIM, int COLS>
__global__ void wmma_gemm_kernel(const _Float16* __restrict__ A,
                                 const _Float16* __restrict__ Bp,
                                 float* __restrict__ C, int Mtiles) {
  constexpr int NT = NDIM / 16;
  constexpr int KT = KDIM / 32;
  __shared__ __align__(16) _Float16 bsh[KT * COLS * 32 * 16];   // 64KB for 8x8

  const int ctile0 = blockIdx.y * COLS;

  // cooperative async stage of this block's B column group
  {
    constexpr int GROUP_BYTES = KT * COLS * 1024;   // per-(kt,ct) fragment = 1KB
    constexpr int CHUNKS = GROUP_BYTES / 16;
    const uint32_t lbase = (uint32_t)(uintptr_t)&bsh[0];
    for (int ci = threadIdx.x; ci < CHUNKS; ci += BLOCK_THREADS) {
      const int lb  = ci * 16;                       // byte offset in group
      const int kt  = lb / (COLS * 1024);
      const int rem = lb - kt * (COLS * 1024);
      const uint64_t g = (uint64_t)(uintptr_t)((const char*)Bp +
                         (size_t)kt * (NT * 1024) + (size_t)ctile0 * 1024 + rem);
      async_copy_b128(lbase + (uint32_t)lb, g);
    }
    wait_asynccnt0();
    __syncthreads();
  }

  const int wave = threadIdx.x >> 5;
  const int lane = threadIdx.x & 31;
  const int rowTile = blockIdx.x * WAVES_PER_BLOCK + wave;
  if (rowTile >= Mtiles) return;                    // wave-uniform, after barrier
  const long row = (long)rowTile * 16 + (lane & 15);
  const int  kb8 = (lane & 16) ? 8 : 0;
  const _Float16* arow = A + row * KDIM;

  v8f acc[COLS] = {};
  #pragma unroll
  for (int k0 = 0; k0 < KDIM; k0 += 32) {
    const _Float16* ap = arow + k0 + kb8;
    v8h lo = *(const v8h*)ap;
    v8h hi = *(const v8h*)(ap + 16);
    v16h a;
    #pragma unroll
    for (int j = 0; j < 8; ++j) { a[j] = lo[j]; a[8 + j] = hi[j]; }
    const int kt = k0 >> 5;
    #pragma unroll
    for (int c = 0; c < COLS; ++c) {
      const v16h b = *(const v16h*)(&bsh[((kt * COLS + c) * 32 + lane) * 16]);
      acc[c] = __builtin_amdgcn_wmma_f32_16x16x32_f16(false, a, false, b,
                                                      (short)0, acc[c], false, false);
    }
  }
  const int  r0      = (lane & 16) ? 8 : 0;
  const long rbase   = (long)rowTile * 16 + r0;
  const int  colLane = lane & 15;
  #pragma unroll
  for (int c = 0; c < COLS; ++c) {
    const int col = (ctile0 + c) * 16 + colLane;
    #pragma unroll
    for (int v = 0; v < 8; ++v)
      C[(rbase + v) * NDIM + col] = acc[c][v];
  }
}

// ------------------------------------------------------- attention kernels

// el[n,h] = sum_f z[n,h,f]*al[h,f] ; er likewise. One wave per node,
// lanes 0-15 cover head 0 (8 feats each), lanes 16-31 head 1.
__global__ void node_scores_kernel(const float* __restrict__ z,
                                   const float* __restrict__ al,
                                   const float* __restrict__ ar,
                                   float* __restrict__ el, float* __restrict__ er, int n) {
  const int wave = threadIdx.x >> 5, lane = threadIdx.x & 31;
  const int node = blockIdx.x * WAVES_PER_BLOCK + wave;
  if (node >= n) return;
  const int head  = lane >> 4;
  const int fbase = (lane & 15) * 8;
  const float* zp  = z  + (long)node * 256 + head * 128 + fbase;
  const float* alp = al + head * 128 + fbase;
  const float* arp = ar + head * 128 + fbase;
  float sl = 0.f, sr = 0.f;
  #pragma unroll
  for (int j = 0; j < 8; ++j) { float v = zp[j]; sl += v * alp[j]; sr += v * arp[j]; }
  #pragma unroll
  for (int m = 1; m < 16; m <<= 1) {
    sl += __shfl_xor(sl, m, 32);
    sr += __shfl_xor(sr, m, 32);
  }
  if ((lane & 15) == 0) { el[node * 2 + head] = sl; er[node * 2 + head] = sr; }
}

__device__ inline void atomicMaxF(float* addr, float val) {
  if (val >= 0.f) atomicMax((int*)addr, __float_as_int(val));
  else            atomicMin((unsigned int*)addr, (unsigned int)__float_as_int(val));
}

__global__ void edge_scores_kernel(const float* __restrict__ el, const float* __restrict__ er,
                                   const int* __restrict__ src, const int* __restrict__ dst,
                                   float* __restrict__ eraw, float* __restrict__ mbuf, long E) {
  long i = (long)blockIdx.x * blockDim.x + threadIdx.x;
  if (i >= E) return;
  const int s = src[i], d = dst[i];
  #pragma unroll
  for (int h = 0; h < 2; ++h) {
    float e = el[s * 2 + h] + er[d * 2 + h];
    e = (e > 0.f) ? e : 0.2f * e;        // leaky relu
    eraw[i * 2 + h] = e;
    atomicMaxF(&mbuf[d * 2 + h], e);
  }
}

__global__ void edge_exp_kernel(float* __restrict__ exb, const float* __restrict__ mbuf,
                                const int* __restrict__ dst, float* __restrict__ ssum, long E) {
  long i = (long)blockIdx.x * blockDim.x + threadIdx.x;
  if (i >= E) return;
  const int d = dst[i];
  #pragma unroll
  for (int h = 0; h < 2; ++h) {
    float m = mbuf[d * 2 + h];
    if (!(m >= -3.0e38f)) m = 0.f;       // isfinite fixup (-inf -> 0)
    float ex = expf(exb[i * 2 + h] - m);
    exb[i * 2 + h] = ex;
    atomicAdd(&ssum[d * 2 + h], ex);
  }
}

// out[dst] += alpha * z[src] ; one wave per edge, coalesced across lanes.
__global__ void edge_aggr_kernel(const float* __restrict__ exb, const float* __restrict__ ssum,
                                 const float* __restrict__ z,
                                 const int* __restrict__ src, const int* __restrict__ dst,
                                 float* __restrict__ agg, long E) {
  const int wave = threadIdx.x >> 5, lane = threadIdx.x & 31;
  const long e = (long)blockIdx.x * WAVES_PER_BLOCK + wave;
  if (e >= E) return;
  const int s = src[e], d = dst[e];
  const float a0 = exb[e * 2 + 0] / ssum[d * 2 + 0];
  const float a1 = exb[e * 2 + 1] / ssum[d * 2 + 1];
  const float* zs = z + (long)s * 256;
  float* o = agg + (long)d * 256;
  #pragma unroll
  for (int j = 0; j < 8; ++j) {
    const int f = j * 32 + lane;
    const float al_ = (f < 128) ? a0 : a1;
    atomicAdd(&o[f], al_ * zs[f]);
  }
}

__global__ void bias_relu_cvt_kernel(const float* __restrict__ agg, const float* __restrict__ b,
                                     _Float16* __restrict__ out16, long n) {
  long i = (long)blockIdx.x * blockDim.x + threadIdx.x;
  if (i >= n) return;
  float v = agg[i] + b[i & 255];
  v = (v > 0.f) ? v : 0.f;
  out16[i] = (_Float16)v;
}

// ------------------------------------------------- fused edge-concat GEMM
// out[e,:] = h2[src[e]] @ Wfc[0:256] + h2[dst[e]] @ Wfc[256:512] + bfc
// 16 edges per wave; A fragments gathered per-lane (no materialized concat).
// Entire packed Wfc (64KB) is staged into LDS via async copies.
__global__ void edge_gemm_kernel(const _Float16* __restrict__ h2,
                                 const _Float16* __restrict__ Wp,   // packed [512x64]
                                 const float* __restrict__ bfc,
                                 const int* __restrict__ src, const int* __restrict__ dst,
                                 float* __restrict__ out, int Etiles) {
  constexpr int KT = 16;                 // K=512 -> 16 k-tiles
  constexpr int NT = 4;                  // N=64  -> 4 col tiles
  __shared__ __align__(16) _Float16 wsh[KT * NT * 32 * 16];       // 64KB

  {
    constexpr int CHUNKS = KT * NT * 1024 / 16;
    const uint32_t lbase = (uint32_t)(uintptr_t)&wsh[0];
    const uint64_t gbase = (uint64_t)(uintptr_t)Wp;
    for (int ci = threadIdx.x; ci < CHUNKS; ci += BLOCK_THREADS) {
      const int lb = ci * 16;
      async_copy_b128(lbase + (uint32_t)lb, gbase + (uint64_t)lb);
    }
    wait_asynccnt0();
    __syncthreads();
  }

  const int wave = threadIdx.x >> 5, lane = threadIdx.x & 31;
  const int rowTile = blockIdx.x * WAVES_PER_BLOCK + wave;
  if (rowTile >= Etiles) return;                  // wave-uniform, after barrier
  const long e_base = (long)rowTile * 16;
  const long myEdge = e_base + (lane & 15);
  const int  kb8    = (lane & 16) ? 8 : 0;
  const _Float16* rowS = h2 + (long)src[myEdge] * 256;
  const _Float16* rowD = h2 + (long)dst[myEdge] * 256;

  v8f acc[4] = {};
  #pragma unroll
  for (int k0 = 0; k0 < 256; k0 += 32) {          // src half (Wfc rows 0..255)
    const _Float16* ap = rowS + k0 + kb8;
    v8h lo = *(const v8h*)ap;
    v8h hi = *(const v8h*)(ap + 16);
    v16h a;
    #pragma unroll
    for (int j = 0; j < 8; ++j) { a[j] = lo[j]; a[8 + j] = hi[j]; }
    const int kt = k0 >> 5;
    #pragma unroll
    for (int c = 0; c < 4; ++c) {
      const v16h b = *(const v16h*)(&wsh[((kt * NT + c) * 32 + lane) * 16]);
      acc[c] = __builtin_amdgcn_wmma_f32_16x16x32_f16(false, a, false, b,
                                                      (short)0, acc[c], false, false);
    }
  }
  #pragma unroll
  for (int k0 = 0; k0 < 256; k0 += 32) {          // dst half (Wfc rows 256..511)
    const _Float16* ap = rowD + k0 + kb8;
    v8h lo = *(const v8h*)ap;
    v8h hi = *(const v8h*)(ap + 16);
    v16h a;
    #pragma unroll
    for (int j = 0; j < 8; ++j) { a[j] = lo[j]; a[8 + j] = hi[j]; }
    const int kt = 8 + (k0 >> 5);
    #pragma unroll
    for (int c = 0; c < 4; ++c) {
      const v16h b = *(const v16h*)(&wsh[((kt * NT + c) * 32 + lane) * 16]);
      acc[c] = __builtin_amdgcn_wmma_f32_16x16x32_f16(false, a, false, b,
                                                      (short)0, acc[c], false, false);
    }
  }
  const int r0 = (lane & 16) ? 8 : 0;
  const int colLane = lane & 15;
  #pragma unroll
  for (int c = 0; c < 4; ++c) {
    const int col  = c * 16 + colLane;
    const float bb = bfc[col];
    #pragma unroll
    for (int v = 0; v < 8; ++v)
      out[(e_base + r0 + v) * 64 + col] = acc[c][v] + bb;
  }
}

// ---------------------------------------------------------------- launcher

extern "C" void kernel_launch(void* const* d_in, const int* in_sizes, int n_in,
                              void* d_out, int out_size, void* d_ws, size_t ws_size,
                              hipStream_t stream) {
  const float* h   = (const float*)d_in[0];
  const int*   src = (const int*)d_in[1];
  const int*   dst = (const int*)d_in[2];
  const float* W1  = (const float*)d_in[4];
  const float* al1 = (const float*)d_in[5];
  const float* ar1 = (const float*)d_in[6];
  const float* b1  = (const float*)d_in[7];
  const float* W2  = (const float*)d_in[8];
  const float* al2 = (const float*)d_in[9];
  const float* ar2 = (const float*)d_in[10];
  const float* b2  = (const float*)d_in[11];
  const float* Wfc = (const float*)d_in[12];
  const float* bfc = (const float*)d_in[13];
  float* out = (float*)d_out;

  const long N = (long)in_sizes[0] / 256;   // 50000
  const long E = (long)in_sizes[1];         // 800000

  // workspace carve-up (256B aligned)
  char* w = (char*)d_ws;
  auto take = [&](size_t bytes) -> char* {
    char* p = w; w += (bytes + 255) & ~(size_t)255; return p;
  };
  _Float16* h16  = (_Float16*)take((size_t)N * 256 * sizeof(_Float16));
  _Float16* w1p  = (_Float16*)take((size_t)256 * 256 * sizeof(_Float16));
  _Float16* w2p  = (_Float16*)take((size_t)256 * 256 * sizeof(_Float16));
  _Float16* wfcp = (_Float16*)take((size_t)512 * 64 * sizeof(_Float16));
  float* z    = (float*)take((size_t)N * 256 * sizeof(float));
  float* agg  = (float*)take((size_t)N * 256 * sizeof(float));
  float* el   = (float*)take((size_t)N * 2 * sizeof(float));
  float* er   = (float*)take((size_t)N * 2 * sizeof(float));
  float* mbuf = (float*)take((size_t)N * 2 * sizeof(float));
  float* ssum = (float*)take((size_t)N * 2 * sizeof(float));
  float* exb  = (float*)take((size_t)E * 2 * sizeof(float));

  const int T = BLOCK_THREADS;
  auto cdiv = [](long a, long b) { return (int)((a + b - 1) / b); };

  // input conversion / weight packing
  cvt_f16_kernel<<<cdiv(N * 256, T), T, 0, stream>>>(h, h16, N * 256);
  pack_b_kernel<256, 256><<<cdiv(256 * 256, T), T, 0, stream>>>(W1, w1p);
  pack_b_kernel<256, 256><<<cdiv(256 * 256, T), T, 0, stream>>>(W2, w2p);
  pack_b_kernel<512, 64><<<cdiv(512 * 64, T), T, 0, stream>>>(Wfc, wfcp);

  const int Mtiles = (int)(N / 16);
  dim3 gemmGrid(cdiv(Mtiles, WAVES_PER_BLOCK), 2);  // 2 groups of 8 col tiles = 256 cols

  for (int layer = 0; layer < 2; ++layer) {
    const _Float16* wp = (layer == 0) ? w1p : w2p;
    const float* al = (layer == 0) ? al1 : al2;
    const float* ar = (layer == 0) ? ar1 : ar2;
    const float* bb = (layer == 0) ? b1  : b2;

    wmma_gemm_kernel<256, 256, 8><<<gemmGrid, T, 0, stream>>>(h16, wp, z, Mtiles);
    node_scores_kernel<<<cdiv(N, WAVES_PER_BLOCK), T, 0, stream>>>(z, al, ar, el, er, (int)N);
    fill_f32_kernel<<<cdiv(N * 2, T), T, 0, stream>>>(mbuf, -INFINITY, N * 2);
    fill_f32_kernel<<<cdiv(N * 2, T), T, 0, stream>>>(ssum, 0.f, N * 2);
    fill_f32_kernel<<<cdiv(N * 256, T), T, 0, stream>>>(agg, 0.f, N * 256);
    edge_scores_kernel<<<cdiv(E, T), T, 0, stream>>>(el, er, src, dst, exb, mbuf, E);
    edge_exp_kernel<<<cdiv(E, T), T, 0, stream>>>(exb, mbuf, dst, ssum, E);
    edge_aggr_kernel<<<cdiv(E, WAVES_PER_BLOCK), T, 0, stream>>>(exb, ssum, z, src, dst, agg, E);
    bias_relu_cvt_kernel<<<cdiv(N * 256, T), T, 0, stream>>>(agg, bb, h16, N * 256);
  }

  edge_gemm_kernel<<<cdiv(E / 16, WAVES_PER_BLOCK), T, 0, stream>>>(
      h16, wfcp, bfc, src, dst, out, (int)(E / 16));
}
